// AttentionMechanism_18459769438864
// MI455X (gfx1250) — matrix operand order repaired and checked by reading, not compiled, not run on hardware
//
#include <hip/hip_runtime.h>
#include <hip/hip_bf16.h>
#include <cstdint>
#include <cstddef>

// Problem sizes (match the reference)
#define B_    128
#define L_    196
#define ENC_  2048
#define DEC_  512
#define ATT_  512
#define ROWS_ (B_ * L_)     // 25088 flattened (b,l) rows; divisible by 16
#define KT_   (ENC_ / 32)   // 64 K-tiles of 32 (bf16 WMMA K)
#define NT_   (ATT_ / 16)   // 32 N-tiles of 16

typedef __attribute__((ext_vector_type(16))) __bf16 v16bf;
typedef __attribute__((ext_vector_type(8)))  float  v8f;

// round-to-nearest-even f32 -> bf16 (bit pattern)
static __device__ __forceinline__ unsigned int f2bf(float x) {
  unsigned int u = __float_as_uint(x);
  return (u + 0x7FFFu + ((u >> 16) & 1u)) >> 16;
}

// Fast tanh: 1 - 2/(e^{2x}+1). v_exp_f32 + v_rcp_f32; saturates to +/-1 at
// extremes (exp overflow -> inf -> 1; underflow -> 0 -> -1).
static __device__ __forceinline__ float tanh_fast(float x) {
  float ex = __expf(2.0f * x);
  return 1.0f - 2.0f / (ex + 1.0f);
}

// ---------------------------------------------------------------------------
// Kernel 1: convert We [ENC, ATT] f32 -> bf16, pre-swizzled into WMMA B-matrix
// fragment layout. Fragment f = nTile*KT_ + kTile is 32 lanes x 16 bf16 = 1KB.
// B (32x16, KxN) contiguous-K order used here: lane = (K>=16 ? 16:0) + n ; element e <-> K%16.
// ---------------------------------------------------------------------------
__global__ __launch_bounds__(256) void k_swizzle_We(const float* __restrict__ We,
                                                    unsigned short* __restrict__ weBf) {
  int t    = blockIdx.x * 256 + threadIdx.x;   // 0 .. 65535 (2048 frags * 32 lanes)
  int lane = t & 31;
  int frag = t >> 5;
  int kTile = frag % KT_;
  int nTile = frag / KT_;
  int n  = nTile * 16 + (lane & 15);
  int k0 = kTile * 32 + (lane >> 4) * 16;      // this lane's contiguous K range

  unsigned int v[16];
  #pragma unroll
  for (int e = 0; e < 16; ++e)
    v[e] = f2bf(We[(size_t)(k0 + e) * ATT_ + n]);

  uint4* dst = (uint4*)(weBf + (size_t)frag * 512 + lane * 16); // ushort units
  uint4 a, b;
  a.x = v[0]  | (v[1]  << 16);  a.y = v[2]  | (v[3]  << 16);
  a.z = v[4]  | (v[5]  << 16);  a.w = v[6]  | (v[7]  << 16);
  b.x = v[8]  | (v[9]  << 16);  b.y = v[10] | (v[11] << 16);
  b.z = v[12] | (v[13] << 16);  b.w = v[14] | (v[15] << 16);
  dst[0] = a; dst[1] = b;
}

// ---------------------------------------------------------------------------
// Kernel 2: dec_map[b][n] = decoder_hidden[b] . Wd[:,n] + bd[n] + be[n]
// (be folded here; ba dropped entirely: softmax(x + ba) == softmax(x))
// ---------------------------------------------------------------------------
__global__ __launch_bounds__(256) void k_dec(const float* __restrict__ dh,
                                             const float* __restrict__ Wd,
                                             const float* __restrict__ bd,
                                             const float* __restrict__ be,
                                             float* __restrict__ dec) {
  __shared__ float sh[DEC_];
  int b = blockIdx.x, t = threadIdx.x;
  for (int i = t; i < DEC_; i += 256) sh[i] = dh[(size_t)b * DEC_ + i];
  __syncthreads();
  for (int n = t; n < ATT_; n += 256) {
    float acc = 0.f;
    #pragma unroll 8
    for (int k = 0; k < DEC_; ++k) acc += sh[k] * Wd[(size_t)k * ATT_ + n];
    dec[(size_t)b * ATT_ + n] = acc + bd[n] + be[n];
  }
}

__global__ __launch_bounds__(256) void k_zero(float* __restrict__ p, int n) {
  int i = blockIdx.x * 256 + threadIdx.x;
  if (i < n) p[i] = 0.f;
}

// ---------------------------------------------------------------------------
// Kernel 3 (main WMMA kernel): scores for 16 flattened rows per block.
// Stage A tile (16 x 2048) as bf16 in LDS using a contiguous-K A (16x32)
// per-lane swizzle. 8 waves; each wave owns 4 N-tiles processed as 2 pairs
// with TWO independent WMMA accumulators in flight (breaks the C->C RAW
// chain, halves LDS A traffic). Fused fast-tanh * wa epilogue, butterfly
// reduce across the 16-lane half, global atomic accumulate.
// ---------------------------------------------------------------------------
__global__ __launch_bounds__(256) void k_scores(const float* __restrict__ enc,
                                                const unsigned short* __restrict__ weBf,
                                                const float* __restrict__ dec,
                                                const float* __restrict__ wa,
                                                float* __restrict__ scores) {
  __shared__ __align__(32) unsigned short As[16 * ENC_];  // 64 KB swizzled bf16 A
  const int tid = threadIdx.x;
  const int r0  = blockIdx.x * 16;

  // ---- Stage A: each thread handles 16 groups of 8 consecutive K values.
  // A (16x32, MxK) contiguous-K order used here: lane = m + 16*khi with khi = (K%32 >= 16),
  // element = K%16; an 8-aligned K group maps to 8 contiguous lane elements.
  #pragma unroll 4
  for (int i = 0; i < 16; ++i) {
    int g_id = i * 256 + tid;
    int m  = g_id >> 8;        // row within tile
    int g  = g_id & 255;       // K-group (8 elements each)
    int K0 = g * 8;
    const float4* src = (const float4*)(enc + (size_t)(r0 + m) * ENC_ + K0);
    float4 f0 = src[0], f1 = src[1];
    int kTile = K0 >> 5;
    int kk    = K0 & 31;
    int khi   = (kk >> 4) & 1;
    int e0    = kk & 8;                   // 0 or 8
    int lane  = m + (khi << 4);
    uint4 w;
    w.x = f2bf(f0.x) | (f2bf(f0.y) << 16);
    w.y = f2bf(f0.z) | (f2bf(f0.w) << 16);
    w.z = f2bf(f1.x) | (f2bf(f1.y) << 16);
    w.w = f2bf(f1.z) | (f2bf(f1.w) << 16);
    *(uint4*)(&As[kTile * 512 + lane * 16 + e0]) = w;
  }
  __syncthreads();

  const int wave = tid >> 5;
  const int lane = tid & 31;
  const int nl   = lane & 15;   // column within N-tile held by this lane (C/D layout)
  const int hi   = lane >> 4;   // C/D: VGPR v holds row m = v + 8*hi

  const float* decRow[8];
  #pragma unroll
  for (int v = 0; v < 8; ++v) {
    int r = r0 + v + 8 * hi;
    decRow[v] = dec + (size_t)(r / L_) * ATT_;   // batch of this row
  }

  float rowAcc[8];
  #pragma unroll
  for (int v = 0; v < 8; ++v) rowAcc[v] = 0.f;

  for (int np = 0; np < 2; ++np) {
    const int nTile0 = wave * 4 + np * 2;        // this pair: nTile0, nTile0+1
    const int n0 = nTile0 * 16 + nl;
    const int n1 = n0 + 16;
    const float wav0 = wa[n0];
    const float wav1 = wa[n1];

    v8f c0 = {}, c1 = {};
    const char* bBase0 = (const char*)weBf + (size_t)nTile0 * KT_ * 1024;
    const char* bBase1 = bBase0 + (size_t)KT_ * 1024;
    #pragma unroll 4
    for (int kt = 0; kt < KT_; ++kt) {
      v16bf a  = *(const v16bf*)(&As[kt * 512 + lane * 16]);
      v16bf b0 = *(const v16bf*)(bBase0 + (size_t)kt * 1024 + lane * 32);
      v16bf b1 = *(const v16bf*)(bBase1 + (size_t)kt * 1024 + lane * 32);
      c0 = __builtin_amdgcn_wmma_f32_16x16x32_bf16(false, a, false, b0,
                                                   (short)0, c0, false, false);
      c1 = __builtin_amdgcn_wmma_f32_16x16x32_bf16(false, a, false, b1,
                                                   (short)0, c1, false, false);
    }

    // Epilogue: per row m = v + 8*hi, sum over both tiles' columns, then
    // butterfly-reduce across the 16-lane half (one reduce for both tiles).
    #pragma unroll
    for (int v = 0; v < 8; ++v) {
      const float* dr = decRow[v];
      float contrib = tanh_fast(c0[v] + dr[n0]) * wav0
                    + tanh_fast(c1[v] + dr[n1]) * wav1;
      contrib += __shfl_xor(contrib, 1, 32);
      contrib += __shfl_xor(contrib, 2, 32);
      contrib += __shfl_xor(contrib, 4, 32);
      contrib += __shfl_xor(contrib, 8, 32);   // stays within 16-lane half
      rowAcc[v] += contrib;
    }
  }

  if (nl == 0) {
    #pragma unroll
    for (int v = 0; v < 8; ++v)
      atomicAdd(&scores[r0 + v + 8 * hi], rowAcc[v]);
  }
}

// ---------------------------------------------------------------------------
// Kernel 4: per-batch softmax over L, write alphas, then context = alphas^T @ enc
// ---------------------------------------------------------------------------
__global__ __launch_bounds__(256) void k_softmax_ctx(const float* __restrict__ scores,
                                                     const float* __restrict__ enc,
                                                     float* __restrict__ out) {
  __shared__ float red[256];
  __shared__ float al[L_];
  const int b = blockIdx.x, t = threadIdx.x;

  float s = (t < L_) ? scores[(size_t)b * L_ + t] : -3.402823466e38f;
  red[t] = s; __syncthreads();
  for (int o = 128; o > 0; o >>= 1) { if (t < o) red[t] = fmaxf(red[t], red[t + o]); __syncthreads(); }
  float mx = red[0]; __syncthreads();

  float e = (t < L_) ? __expf(s - mx) : 0.f;
  red[t] = e; __syncthreads();
  for (int o = 128; o > 0; o >>= 1) { if (t < o) red[t] += red[t + o]; __syncthreads(); }
  float inv = 1.f / red[0];

  float a = e * inv;
  if (t < L_) {
    al[t] = a;
    out[(size_t)B_ * ENC_ + (size_t)b * L_ + t] = a;   // alphas after context
  }
  __syncthreads();

  const float* eb = enc + (size_t)b * L_ * ENC_;
  float acc[ENC_ / 256];
  #pragma unroll
  for (int j = 0; j < ENC_ / 256; ++j) acc[j] = 0.f;
  for (int l = 0; l < L_; ++l) {
    const float* row = eb + (size_t)l * ENC_;
    const float av = al[l];
    #pragma unroll
    for (int j = 0; j < ENC_ / 256; ++j) acc[j] += av * row[t + j * 256];
  }
  #pragma unroll
  for (int j = 0; j < ENC_ / 256; ++j)
    out[(size_t)b * ENC_ + t + j * 256] = acc[j];
}

// ---------------------------------------------------------------------------
extern "C" void kernel_launch(void* const* d_in, const int* in_sizes, int n_in,
                              void* d_out, int out_size, void* d_ws, size_t ws_size,
                              hipStream_t stream) {
  (void)in_sizes; (void)n_in; (void)out_size; (void)ws_size;
  const float* enc = (const float*)d_in[0];
  const float* dh  = (const float*)d_in[1];
  const float* We  = (const float*)d_in[2];
  const float* be  = (const float*)d_in[3];
  const float* Wd  = (const float*)d_in[4];
  const float* bd  = (const float*)d_in[5];
  const float* wa  = (const float*)d_in[6];
  // d_in[7] = ba: uniform shift of scores -> softmax-invariant -> unused.
  float* out = (float*)d_out;

  // Workspace layout (~2.35 MB total)
  unsigned short* weBf  = (unsigned short*)d_ws;                                // 2 MB
  float*          dec   = (float*)((char*)d_ws + 2u * 1024 * 1024);             // 256 KB
  float*          score = (float*)((char*)d_ws + 2u * 1024 * 1024 + 256u * 1024); // 100 KB

  k_swizzle_We<<<dim3(256), dim3(256), 0, stream>>>(We, weBf);
  k_dec<<<dim3(B_), dim3(256), 0, stream>>>(dh, Wd, bd, be, dec);
  k_zero<<<dim3((ROWS_ + 255) / 256), dim3(256), 0, stream>>>(score, ROWS_);
  k_scores<<<dim3(ROWS_ / 16), dim3(256), 0, stream>>>(enc, weBf, dec, wa, score);
  k_softmax_ctx<<<dim3(B_), dim3(256), 0, stream>>>(score, enc, out);
}
